// LSTM_19602230739967
// MI455X (gfx1250) — matrix operand order repaired and checked by reading, compiled, not run on hardware
//
#include <hip/hip_runtime.h>

typedef __attribute__((ext_vector_type(16))) __bf16 v16bf;
typedef __attribute__((ext_vector_type(8)))  __bf16 v8bf;
typedef __attribute__((ext_vector_type(8)))  float  v8f;

#define HD     512      // hidden size
#define IDIM   1024     // input feature dim
#define TSTEPS 512
#define BREAL  50
#define BPAD   64       // padded batch rows
#define NWG    64       // persistent workgroups in scan (each owns 8 hidden cols)

// ---------------- workspace layout (bytes) ----------------
#define OFF_BAR    ((size_t)0)                          // 4096
#define OFF_HBUF   ((size_t)4096)                       // 2*64*512 bf16 = 131072
#define OFF_HLAST  (OFF_HBUF + 131072)                  // 64*512 f32   = 131072
#define OFF_X      (OFF_HLAST + 131072)                 // (T,64,H) bf16 = 33554432
#define OFF_HS0    (OFF_X + 33554432)                   // (T,64,H) bf16 = 33554432
#define OFF_INBF   (OFF_HS0 + 33554432)                 // 50*512*1024 bf16 = 52428800
#define OFF_PCAW   (OFF_INBF + 52428800)                // 512*1024 bf16 = 1048576

__device__ __forceinline__ float sigm(float x) { return 1.f / (1.f + __expf(-x)); }

// Build a 16x32 bf16 A-fragment (per ISA 7.12.2 16-bit A layout):
// lane row = lane%16; elems 0..7 at k0 + (lane/16)*8, elems 8..15 at k0+16+(lane/16)*8
__device__ __forceinline__ v16bf load_afrag(const __bf16* rowbase_k0, int khalf) {
  v8bf lo = *(const v8bf*)(rowbase_k0 + khalf * 8);
  v8bf hi = *(const v8bf*)(rowbase_k0 + 16 + khalf * 8);
  v16bf a;
#pragma unroll
  for (int i = 0; i < 8; ++i) { a[i] = lo[i]; a[i + 8] = hi[i]; }
  return a;
}

// ---------------- f32 -> bf16 convert ----------------
__global__ void f32_to_bf16(const float* __restrict__ in, __bf16* __restrict__ out, int n) {
  int i = blockIdx.x * blockDim.x + threadIdx.x;
  int stride = gridDim.x * blockDim.x;
  for (; i < n; i += stride) out[i] = (__bf16)in[i];
}

// ---------------- PCA GEMM: x = input @ pca_W.T + pca_b ----------------
// M = B*T = 25600 (b-major), N = 512, K = 1024. Output layout (T, 64pad, H) bf16,
// padded rows pre-zeroed by memset. grid(400,4), block 256 (8 waves); wave tile 64x16.
__global__ __launch_bounds__(256) void pca_gemm(const __bf16* __restrict__ A,
                                                const __bf16* __restrict__ Wb,
                                                const float* __restrict__ bias,
                                                __bf16* __restrict__ xout) {
  const int w = threadIdx.x >> 5, lane = threadIdx.x & 31;
  const int khalf = lane >> 4, lrow = lane & 15;
  const int M0 = blockIdx.x * 64;
  const int n  = blockIdx.y * 128 + w * 16 + lrow;   // B-fragment col & C col for this lane

  v8f acc[4];
#pragma unroll
  for (int m = 0; m < 4; ++m) acc[m] = (v8f){0.f,0.f,0.f,0.f,0.f,0.f,0.f,0.f};

  for (int kb = 0; kb < 32; ++kb) {
    const int k0 = kb * 32;
    // B fragment: elems 0..7 at k0 + khalf*8, elems 8..15 at k0 + 16 + khalf*8
    v16bf b = load_afrag(Wb + (size_t)n * IDIM + k0, khalf);
#pragma unroll
    for (int mt = 0; mt < 4; ++mt) {
      const int m = M0 + mt * 16 + lrow;
      v16bf a = load_afrag(A + (size_t)m * IDIM + k0, khalf);
      acc[mt] = __builtin_amdgcn_wmma_f32_16x16x32_bf16(false, a, false, b, (short)0,
                                                        acc[mt], false, false);
    }
  }
  const float bv = bias[n];
#pragma unroll
  for (int mt = 0; mt < 4; ++mt) {
#pragma unroll
    for (int v = 0; v < 8; ++v) {
      const int row = M0 + mt * 16 + v + 8 * khalf;   // row = b*T + t
      const int bi = row >> 9;                        // T == 512
      const int t  = row & 511;
      xout[((size_t)t * BPAD + bi) * HD + n] = (__bf16)(acc[mt][v] + bv);
    }
  }
}

// ---------------- persistent fused LSTM scan (both layers) ----------------
__device__ __forceinline__ void grid_bar(unsigned* bar, unsigned epoch) {
  __syncthreads();
  if (threadIdx.x == 0) {
    __threadfence();
    atomicAdd(bar, 1u);
    const unsigned target = epoch * NWG;
    while (atomicAdd(bar, 0u) < target) { __builtin_amdgcn_s_sleep(2); }
  }
  __syncthreads();
  __threadfence();
}

__global__ __launch_bounds__(256) void lstm_scan(
    const float* __restrict__ W_ih0, const float* __restrict__ W_hh0, const float* __restrict__ b0,
    const float* __restrict__ W_ih1, const float* __restrict__ W_hh1, const float* __restrict__ b1,
    const float* __restrict__ h0, const float* __restrict__ c0,
    const __bf16* __restrict__ x0,   // (T,64,H) layer-0 input (PCA output)
    __bf16* __restrict__ hs0,        // (T,64,H) layer-0 output / layer-1 input
    __bf16* __restrict__ hbuf,       // 2 x 64 x 512 bf16 ping-pong h
    float* __restrict__ hlast,       // 64 x 512 f32 final h of layer 1
    unsigned* __restrict__ bar) {
  // Pre-swizzled combined weights [Wih;Whh] for this WG's 32 gate cols:
  // ldsW[nt(2)][kb(32)][lane(32)][e(16)] bf16 = 64 KB
  __shared__ __bf16 ldsW[2 * 32 * 32 * 16];
  __shared__ float  ldsG[BPAD * 32];   // gate pre-activations (64 rows x 32 gate cols)
  __shared__ float  ldsC[BPAD * 8];    // c-state for this WG's 8 hidden cols

  const int tid = threadIdx.x;
  const int wg  = blockIdx.x;
  const int hc0 = wg * 8;              // first hidden column owned by this WG
  const int w = tid >> 5, lane = tid & 31;
  const int mt = w >> 1, nt = w & 1;   // wave tile: rows mt*16.., gate-col tile nt
  const int khalf = lane >> 4, lrow = lane & 15;
  unsigned epoch = 0;

  for (int layer = 0; layer < 2; ++layer) {
    const float* Wih  = layer ? W_ih1 : W_ih0;
    const float* Whh  = layer ? W_hh1 : W_hh0;
    const float* bias = layer ? b1 : b0;
    const __bf16* xsrc = layer ? hs0 : x0;

    // ---- fill LDS weights (B-fragment layout) ----
    __syncthreads();
    for (int idx = tid; idx < 2 * 32 * 32 * 16; idx += 256) {
      const int e   = idx & 15;
      const int ln  = (idx >> 4) & 31;
      const int kb  = (idx >> 9) & 31;
      const int ntl = idx >> 14;
      const int k = kb * 32 + ((e >> 3) << 4) + ((ln >> 4) << 3) + (e & 7);        // K index in [0,1024)
      const int j = ntl * 16 + (ln & 15);                   // local gate col 0..31
      const int gcol = ((j >> 3) << 9) + hc0 + (j & 7);     // gate group * 512 + hidden col
      const float v = (k < HD) ? Wih[(size_t)gcol * HD + k]
                               : Whh[(size_t)gcol * HD + (k - HD)];
      ldsW[idx] = (__bf16)v;
    }
    // ---- init c (LDS) and h (hbuf[0], own columns) ----
    for (int idx = tid; idx < BPAD * 8; idx += 256) {
      const int row = idx >> 3, j2 = idx & 7;
      float cv = 0.f, hv = 0.f;
      if (row < BREAL) {
        cv = c0[(size_t)layer * BREAL * HD + row * HD + hc0 + j2];
        hv = h0[(size_t)layer * BREAL * HD + row * HD + hc0 + j2];
      }
      ldsC[idx] = cv;
      hbuf[row * HD + hc0 + j2] = (__bf16)hv;
    }
    epoch++; grid_bar(bar, epoch);

    // bias for this lane's gate column (C col = lane%16 within tile nt)
    const int jl = nt * 16 + lrow;
    const float bval = bias[((jl >> 3) << 9) + hc0 + (jl & 7)];

    int cur = 0;
    for (int t = 0; t < TSTEPS; ++t) {
      v8f acc = (v8f){0.f,0.f,0.f,0.f,0.f,0.f,0.f,0.f};
      const __bf16* xrow = xsrc + (size_t)t * BPAD * HD;
      const __bf16* hb   = hbuf + (size_t)cur * BPAD * HD;
      const int m = mt * 16 + lrow;
#pragma unroll 4
      for (int kb = 0; kb < 32; ++kb) {
        const int k0 = kb * 32;
        const __bf16* asrc = (kb < 16) ? (xrow + (size_t)m * HD + k0)
                                       : (hb + (size_t)m * HD + (k0 - HD));
        v16bf a = load_afrag(asrc, khalf);
        v16bf b = *(const v16bf*)&ldsW[((nt * 32 + kb) * 32 + lane) * 16];
        acc = __builtin_amdgcn_wmma_f32_16x16x32_bf16(false, a, false, b, (short)0,
                                                      acc, false, false);
      }
      // stash gate pre-activations (+bias) to LDS
#pragma unroll
      for (int v = 0; v < 8; ++v) {
        const int row = mt * 16 + v + 8 * khalf;
        ldsG[row * 32 + nt * 16 + lrow] = acc[v] + bval;
      }
      __syncthreads();
      // elementwise gate update for this WG's 8 hidden cols
      for (int idx = tid; idx < BPAD * 8; idx += 256) {
        const int row = idx >> 3, j2 = idx & 7;
        const float gi = ldsG[row * 32 + j2];
        const float gf = ldsG[row * 32 + 8 + j2];
        const float gg = ldsG[row * 32 + 16 + j2];
        const float go = ldsG[row * 32 + 24 + j2];
        float c = ldsC[idx];
        c = sigm(gf) * c + sigm(gi) * tanhf(gg);
        const float h = sigm(go) * tanhf(c);
        ldsC[idx] = c;
        const __bf16 h16 = (__bf16)h;
        hbuf[(size_t)(cur ^ 1) * BPAD * HD + row * HD + hc0 + j2] = h16;
        if (layer == 0) hs0[(size_t)t * BPAD * HD + row * HD + hc0 + j2] = h16;
        else if (t == TSTEPS - 1) hlast[row * HD + hc0 + j2] = h;
      }
      epoch++; grid_bar(bar, epoch);
      cur ^= 1;
    }
  }
}

// ---------------- head: out = tanh(h@W1.T+b1) @ W2.T + b2 ----------------
__global__ void head(const float* __restrict__ hlast,
                     const float* __restrict__ W1, const float* __restrict__ b1,
                     const float* __restrict__ W2, const float* __restrict__ b2,
                     float* __restrict__ out) {
  __shared__ float t1[BREAL * 10];
  const int tid = threadIdx.x;
  if (tid < BREAL * 10) {
    const int b = tid / 10, j = tid % 10;
    float s = b1[j];
    for (int k = 0; k < HD; ++k) s += hlast[b * HD + k] * W1[j * HD + k];
    t1[b * 10 + j] = tanhf(s);
  }
  __syncthreads();
  if (tid < BREAL * 2) {
    const int b = tid / 2, o = tid % 2;
    float s = b2[o];
    for (int j = 0; j < 10; ++j) s += t1[b * 10 + j] * W2[o * 10 + j];
    out[b * 2 + o] = s;
  }
}

extern "C" void kernel_launch(void* const* d_in, const int* in_sizes, int n_in,
                              void* d_out, int out_size, void* d_ws, size_t ws_size,
                              hipStream_t stream) {
  (void)in_sizes; (void)n_in; (void)out_size; (void)ws_size;
  const float* input  = (const float*)d_in[0];
  const float* pca_W  = (const float*)d_in[1];
  const float* pca_b  = (const float*)d_in[2];
  const float* W_ih0  = (const float*)d_in[3];
  const float* W_hh0  = (const float*)d_in[4];
  const float* b0     = (const float*)d_in[5];
  const float* W_ih1  = (const float*)d_in[6];
  const float* W_hh1  = (const float*)d_in[7];
  const float* b1     = (const float*)d_in[8];
  const float* h0     = (const float*)d_in[9];
  const float* c0     = (const float*)d_in[10];
  const float* lin1_W = (const float*)d_in[11];
  const float* lin1_b = (const float*)d_in[12];
  const float* lin2_W = (const float*)d_in[13];
  const float* lin2_b = (const float*)d_in[14];

  char* ws = (char*)d_ws;
  unsigned* bar  = (unsigned*)(ws + OFF_BAR);
  __bf16* hbuf   = (__bf16*)(ws + OFF_HBUF);
  float*  hlast  = (float*)(ws + OFF_HLAST);
  __bf16* x_ws   = (__bf16*)(ws + OFF_X);
  __bf16* hs0    = (__bf16*)(ws + OFF_HS0);
  __bf16* in_bf  = (__bf16*)(ws + OFF_INBF);
  __bf16* pW_bf  = (__bf16*)(ws + OFF_PCAW);

  // reset barrier, zero padded rows of PCA output (graph-capture-safe memsets)
  hipMemsetAsync(bar, 0, 4096, stream);
  hipMemsetAsync(x_ws, 0, (size_t)TSTEPS * BPAD * HD * sizeof(__bf16), stream);

  f32_to_bf16<<<2048, 256, 0, stream>>>(input, in_bf, BREAL * TSTEPS * IDIM);
  f32_to_bf16<<<512, 256, 0, stream>>>(pca_W, pW_bf, HD * IDIM);

  pca_gemm<<<dim3(400, 4), 256, 0, stream>>>(in_bf, pW_bf, pca_b, x_ws);

  lstm_scan<<<NWG, 256, 0, stream>>>(W_ih0, W_hh0, b0, W_ih1, W_hh1, b1,
                                     h0, c0, x_ws, hs0, hbuf, hlast, bar);

  head<<<1, 512, 0, stream>>>(hlast, lin1_W, lin1_b, lin2_W, lin2_b, (float*)d_out);
}
